// ImprovedActorCriticNetwork_82815559402267
// MI455X (gfx1250) — matrix operand order repaired and checked by reading, not compiled, not run on hardware
//
#include <hip/hip_runtime.h>
#include <hip/hip_bf16.h>

// ---------------- problem constants (from reference) ----------------
#define N_MISSIONS 1024
#define N_UAVS     16
#define B_GRAPHS   64
#define EMBED      64
#define HIDDEN     128
#define HEADS      8
#define NTOT       (N_MISSIONS * N_UAVS)   // 16384
#define N_EDGES    (NTOT * 16)             // 262144
#define HC         (HEADS * EMBED)         // 512

typedef __attribute__((ext_vector_type(16))) _Float16 v16h;
typedef __attribute__((ext_vector_type(8)))  float    v8f;

// ======================================================================
// WMMA GEMM:  C[M,N] = act( A[M,K](f16) * Bpacked[K,N](f16) + bias[N] )
// One wave computes NT consecutive 16x16 tiles (16 x NT*16 output strip),
// reusing the A fragment across NT v_wmma_f32_16x16x32_f16 issues.
//
// A fragment (ISA 7.12.2, 16-bit A 16x32): lane<16 rows M=0..15,
//   halves 0..7 = K 0..7, halves 8..15 = K 16..23; lanes>=16 same rows,
//   K 8..15 / 24..31  -> two contiguous 8-half runs = 2x global_load_b128.
// B is PRE-PACKED by pack_w_kernel into fragment order:
//   packed[((k/32)*N + col)*32 + (k%32)] = B[k, col]
//   lane l (col = tile+l&15) reads two 8-half runs at
//   ((kt*N+col)*32 + (l>=16)*8) and +16  -> 2x global_load_b128, 16B aligned.
// C/D layout: VGPR r, lane l -> M = r + 8*(l>=16), N = l&15.
// ======================================================================
template <int NT>
__global__ void wmma_gemm_kernel(const _Float16* __restrict__ A,
                                 const _Float16* __restrict__ Bp,
                                 const float* __restrict__ bias,
                                 float* __restrict__ C,
                                 int M, int N, int K, int relu)
{
    const int wave   = threadIdx.x >> 5;
    const int lane   = threadIdx.x & 31;
    const int m0     = blockIdx.x * 16;
    const int ntile0 = (blockIdx.y * (blockDim.x >> 5) + wave) * NT;
    if (ntile0 * 16 >= N) return;              // wave-uniform, EXEC stays full
    const int lhalf  = lane >> 4;              // 0 | 1
    const int l15    = lane & 15;
    const int arow   = m0 + l15;

    v8f acc[NT];
#pragma unroll
    for (int j = 0; j < NT; ++j) acc[j] = (v8f){};

    const int nkt = K >> 5;
    for (int kt = 0; kt < nkt; ++kt) {
        const int k0 = kt << 5;
        const _Float16* Ap = A + (size_t)arow * K + k0 + (lhalf ? 8 : 0);
        if (kt + 1 < nkt) {                     // gfx1250 global_prefetch_b8
            __builtin_prefetch(Ap + 32, 0, 1);
            __builtin_prefetch(Bp + ((size_t)(kt + 1) * N + ntile0 * 16 + l15) * 32, 0, 1);
        }
        v16h a;
#pragma unroll
        for (int h = 0; h < 8; ++h) { a[h] = Ap[h]; a[h + 8] = Ap[h + 16]; }
#pragma unroll
        for (int j = 0; j < NT; ++j) {
            const int bcol = (ntile0 + j) * 16 + l15;
            const _Float16* Bq = Bp + ((size_t)kt * N + bcol) * 32 + lhalf * 8;
            v16h b;
#pragma unroll
            for (int h = 0; h < 8; ++h) { b[h] = Bq[h]; b[h + 8] = Bq[h + 16]; }
            acc[j] = __builtin_amdgcn_wmma_f32_16x16x32_f16(
                         false, a, false, b, (short)0, acc[j], false, false);
        }
    }

#pragma unroll
    for (int j = 0; j < NT; ++j) {
        const int bcol = (ntile0 + j) * 16 + l15;
        const float bb = bias ? bias[bcol] : 0.0f;
#pragma unroll
        for (int r = 0; r < 8; ++r) {
            float v = acc[j][r] + bb;
            if (relu) v = fmaxf(v, 0.0f);
            C[(size_t)(m0 + r + lhalf * 8) * N + bcol] = v;
        }
    }
}

// ---------------- feature build: x[NTOT,6] ----------------
__global__ void build_feat_kernel(const float* __restrict__ mc,
                                  const int*  __restrict__ mask,
                                  const float* __restrict__ speeds,
                                  const float* __restrict__ cm,
                                  const float* __restrict__ at,
                                  float* __restrict__ feat)
{
    int i = blockIdx.x * blockDim.x + threadIdx.x;
    if (i >= NTOT) return;
    int m = i >> 4, u = i & 15;
    float* f = feat + (size_t)i * 6;
    f[0] = mc[2 * m];      f[1] = mc[2 * m + 1];
    f[2] = (float)mask[m]; f[3] = speeds[u];
    f[4] = cm[u * N_MISSIONS + m];
    f[5] = at[u * N_MISSIONS + m];
}

// ---------------- layer-1 tiny-K GEMM (K=6) ----------------
__global__ void gemm6_kernel(const float* __restrict__ feat,
                             const float* __restrict__ W,
                             const float* __restrict__ bias,
                             float* __restrict__ out)
{
    int idx = blockIdx.x * blockDim.x + threadIdx.x;    // NTOT*512
    if (idx >= NTOT * HC) return;
    int row = idx / HC, col = idx - row * HC;
    const float* f = feat + (size_t)row * 6;
    float acc = bias[col];
#pragma unroll
    for (int k = 0; k < 6; ++k) acc += f[k] * W[k * HC + col];
    out[idx] = acc;
}

// ---------------- fills / converts ----------------
__global__ void fill_u32_kernel(unsigned* p, unsigned v, int n)
{ int i = blockIdx.x * blockDim.x + threadIdx.x; if (i < n) p[i] = v; }

__global__ void fill_f32_kernel(float* p, float v, int n)
{ int i = blockIdx.x * blockDim.x + threadIdx.x; if (i < n) p[i] = v; }

__global__ void to_half_kernel(const float* __restrict__ s, _Float16* __restrict__ d, int n)
{ int i = blockIdx.x * blockDim.x + threadIdx.x; if (i < n) d[i] = (_Float16)s[i]; }

__global__ void relu_to_half_kernel(const float* __restrict__ s, _Float16* __restrict__ d, int n)
{ int i = blockIdx.x * blockDim.x + threadIdx.x; if (i < n) d[i] = (_Float16)fmaxf(s[i], 0.0f); }

// pack weights into WMMA B-fragment order, zero-padding rows >= K0:
//   dst[((k/32)*N + c)*32 + (k%32)] = (k < K0) ? (f16)w[k,c] : 0
__global__ void pack_w_kernel(const float* __restrict__ w, _Float16* __restrict__ d,
                              int K0, int Kp, int N)
{
    int i = blockIdx.x * blockDim.x + threadIdx.x;   // over Kp*N (Kp % 32 == 0)
    if (i >= Kp * N) return;
    int k = i / N, c = i - k * N;
    int kt = k >> 5, kk = k & 31;
    d[((size_t)kt * N + c) * 32 + kk] = (k < K0) ? (_Float16)w[(size_t)k * N + c]
                                                 : (_Float16)0.0f;
}

// ---------------- edge attention (segment softmax + scatter) ----------------
__device__ __forceinline__ unsigned fenc(float f) {
    unsigned u = __float_as_uint(f);
    return (u & 0x80000000u) ? ~u : (u | 0x80000000u);
}
__device__ __forceinline__ float fdec(unsigned u) {
    u = (u & 0x80000000u) ? (u & 0x7FFFFFFFu) : ~u;
    return __uint_as_float(u);
}

__global__ void edge_alpha_kernel(const int* __restrict__ ei,
                                  const float* __restrict__ q,
                                  const float* __restrict__ k,
                                  float* __restrict__ alpha,
                                  unsigned* __restrict__ mEnc)
{
    int t = blockIdx.x * blockDim.x + threadIdx.x;      // E * HEADS
    if (t >= N_EDGES * HEADS) return;
    int e = t >> 3, h = t & 7;
    int src = ei[e], dst = ei[N_EDGES + e];
    const float4* qp = (const float4*)(q + (size_t)dst * HC + h * EMBED);
    const float4* kp = (const float4*)(k + (size_t)src * HC + h * EMBED);
    float s = 0.0f;
#pragma unroll
    for (int c = 0; c < EMBED / 4; ++c) {
        float4 qa = qp[c], ka = kp[c];
        s += qa.x * ka.x + qa.y * ka.y + qa.z * ka.z + qa.w * ka.w;
    }
    s *= 0.125f;                                        // 1/sqrt(EMBED)
    alpha[t] = s;
    atomicMax(mEnc + (size_t)dst * HEADS + h, fenc(s));
}

__global__ void edge_exp_kernel(const int* __restrict__ ei,
                                float* __restrict__ alpha,
                                const unsigned* __restrict__ mEnc,
                                float* __restrict__ den)
{
    int t = blockIdx.x * blockDim.x + threadIdx.x;
    if (t >= N_EDGES * HEADS) return;
    int e = t >> 3, h = t & 7;
    int dst = ei[N_EDGES + e];
    float m  = fdec(mEnc[(size_t)dst * HEADS + h]);
    float ex = __expf(alpha[t] - m);
    alpha[t] = ex;
    atomicAdd(den + (size_t)dst * HEADS + h, ex);
}

__global__ void edge_scatter_kernel(const int* __restrict__ ei,
                                    const float* __restrict__ alpha,
                                    const float* __restrict__ den,
                                    const float* __restrict__ v,
                                    float* __restrict__ sbuf)
{
    int t = blockIdx.x * blockDim.x + threadIdx.x;
    if (t >= N_EDGES * HEADS) return;
    int e = t >> 3, h = t & 7;
    int src = ei[e], dst = ei[N_EDGES + e];
    float a = alpha[t] / (den[(size_t)dst * HEADS + h] + 1e-16f);
    const float* vp = v + (size_t)src * HC + h * EMBED;
    float* op = sbuf + (size_t)dst * HC + h * EMBED;
#pragma unroll 4
    for (int c = 0; c < EMBED; ++c) atomicAdd(op + c, a * vp[c]);
}

// ---------------- per-graph dense MHA (n=256, hd=8) ----------------
__global__ void mha_kernel(const float* __restrict__ qkv, float* __restrict__ o)
{
    __shared__ float sq[256][8], sk[256][8], sv[256][8];
    int b = blockIdx.x, h = blockIdx.y, i = threadIdx.x;
    int row = b * 256 + i;
    const float* base = qkv + (size_t)row * (3 * EMBED);
#pragma unroll
    for (int c = 0; c < 8; ++c) {
        sq[i][c] = base[h * 8 + c];
        sk[i][c] = base[EMBED + h * 8 + c];
        sv[i][c] = base[2 * EMBED + h * 8 + c];
    }
    __syncthreads();
    const float sc = 0.3535533905932738f;               // 1/sqrt(8)
    float mx = -INFINITY, sum = 0.0f, acc[8] = {};
    for (int j = 0; j < 256; ++j) {
        float s = 0.0f;
#pragma unroll
        for (int c = 0; c < 8; ++c) s += sq[i][c] * sk[j][c];
        s *= sc;
        float nm = fmaxf(mx, s);
        float corr = __expf(mx - nm);
        float w = __expf(s - nm);
        sum = sum * corr + w;
#pragma unroll
        for (int c = 0; c < 8; ++c) acc[c] = acc[c] * corr + w * sv[j][c];
        mx = nm;
    }
    float* op = o + (size_t)row * EMBED + h * 8;
#pragma unroll
    for (int c = 0; c < 8; ++c) op[c] = acc[c] / sum;
}

// ---------------- layer norm over 64 ----------------
__global__ void layernorm_kernel(const float* __restrict__ in,
                                 const float* __restrict__ g,
                                 const float* __restrict__ b,
                                 float* __restrict__ out)
{
    int r = blockIdx.x * blockDim.x + threadIdx.x;
    if (r >= NTOT) return;
    const float* x = in + (size_t)r * EMBED;
    float mu = 0.0f;
#pragma unroll 8
    for (int c = 0; c < EMBED; ++c) mu += x[c];
    mu *= (1.0f / EMBED);
    float var = 0.0f;
#pragma unroll 8
    for (int c = 0; c < EMBED; ++c) { float d0 = x[c] - mu; var += d0 * d0; }
    var *= (1.0f / EMBED);
    float inv = rsqrtf(var + 1e-5f);
    float* y = out + (size_t)r * EMBED;
#pragma unroll 8
    for (int c = 0; c < EMBED; ++c) y[c] = (x[c] - mu) * inv * g[c] + b[c];
}

// ---------------- combined head input (faithful torch index layouts) ---------
// row r (uav-major): [uavs_info[r%16], me = ln[(r%1024)*16 + r/1024], speeds[r%16], 0-pad]
__global__ void build_comb_kernel(const float* __restrict__ ln,
                                  const float* __restrict__ uavs,
                                  const float* __restrict__ speeds,
                                  _Float16* __restrict__ comb)   // [NTOT, 96]
{
    int r = blockIdx.x * blockDim.x + threadIdx.x;
    if (r >= NTOT) return;
    int u16 = r & 15;
    int m = r & (N_MISSIONS - 1);
    int u = r >> 10;
    const float* me = ln + (size_t)(m * N_UAVS + u) * EMBED;
    _Float16* c = comb + (size_t)r * 96;
    c[0] = (_Float16)uavs[2 * u16];
    c[1] = (_Float16)uavs[2 * u16 + 1];
#pragma unroll 8
    for (int j = 0; j < EMBED; ++j) c[2 + j] = (_Float16)me[j];
    c[66] = (_Float16)speeds[u16];
#pragma unroll
    for (int j = 67; j < 96; ++j) c[j] = (_Float16)0.0f;
}

// ---------------- row softmax over 1024 (in place on d_out) ----------------
__global__ void softmax_rows_kernel(float* __restrict__ x)
{
    __shared__ float red[256];
    int r = blockIdx.x, t = threadIdx.x;
    float* p = x + (size_t)r * N_MISSIONS;
    float mx = -INFINITY;
    for (int j = t; j < N_MISSIONS; j += 256) mx = fmaxf(mx, p[j]);
    red[t] = mx; __syncthreads();
    for (int s = 128; s > 0; s >>= 1) { if (t < s) red[t] = fmaxf(red[t], red[t + s]); __syncthreads(); }
    mx = red[0]; __syncthreads();
    float sum = 0.0f;
    for (int j = t; j < N_MISSIONS; j += 256) { float e = __expf(p[j] - mx); p[j] = e; sum += e; }
    red[t] = sum; __syncthreads();
    for (int s = 128; s > 0; s >>= 1) { if (t < s) red[t] += red[t + s]; __syncthreads(); }
    float inv = 1.0f / red[0];
    for (int j = t; j < N_MISSIONS; j += 256) p[j] *= inv;
}

// ---------------- critic output (N=1) ----------------
__global__ void critic_out_kernel(const float* __restrict__ hc,
                                  const float* __restrict__ w2,
                                  const float* __restrict__ b2,
                                  float* __restrict__ out)
{
    int r = blockIdx.x * blockDim.x + threadIdx.x;
    if (r >= NTOT) return;
    const float* h = hc + (size_t)r * HIDDEN;
    float acc = b2[0];
#pragma unroll 8
    for (int c = 0; c < HIDDEN; ++c) acc += h[c] * w2[c];
    out[r] = acc;
}

// ======================================================================
// host-side orchestration
// ======================================================================
static inline int cdiv(int a, int b) { return (a + b - 1) / b; }

extern "C" void kernel_launch(void* const* d_in, const int* in_sizes, int n_in,
                              void* d_out, int out_size, void* d_ws, size_t ws_size,
                              hipStream_t stream)
{
    // ---- inputs (setup_inputs() dict order, params depth-first) ----
    const float* mc     = (const float*)d_in[0];
    const int*   ei     = (const int*)  d_in[1];
    /* batch d_in[2] unused */
    const float* uavs   = (const float*)d_in[3];
    const int*   amask  = (const int*)  d_in[4];
    const float* speeds = (const float*)d_in[5];
    const float* cm     = (const float*)d_in[6];
    const float* at     = (const float*)d_in[7];
    // convs[l]: wq,bq,wk,bk,wv,bv,ws,bs at 8 + 8*l
    const float* gnn_w = (const float*)d_in[40]; const float* gnn_b = (const float*)d_in[41];
    const float* ipw   = (const float*)d_in[42]; const float* ipb   = (const float*)d_in[43];
    const float* opw   = (const float*)d_in[44]; const float* opb   = (const float*)d_in[45];
    const float* lng   = (const float*)d_in[46]; const float* lnb   = (const float*)d_in[47];
    const float* aw1   = (const float*)d_in[48]; const float* ab1   = (const float*)d_in[49];
    const float* aw2   = (const float*)d_in[50]; const float* ab2   = (const float*)d_in[51];
    const float* cw1   = (const float*)d_in[52]; const float* cb1   = (const float*)d_in[53];
    const float* cw2   = (const float*)d_in[54]; const float* cb2   = (const float*)d_in[55];

    // ---- workspace carving ----
    char* ws = (char*)d_ws;
    size_t off = 0;
    auto carve = [&](size_t bytes) { size_t r = off; off = (off + bytes + 255) & ~(size_t)255; return (void*)(ws + r); };
    float*    feat  = (float*)   carve((size_t)NTOT * 6 * 4);
    float*    sbuf  = (float*)   carve((size_t)NTOT * HC * 4);   // s-branch + aggregation
    _Float16* xh    = (_Float16*)carve((size_t)NTOT * HC * 2);   // f16 activations (GEMM A)
    float*    qb    = (float*)   carve((size_t)NTOT * HC * 4);
    float*    kb    = (float*)   carve((size_t)NTOT * HC * 4);
    float*    vb    = (float*)   carve((size_t)NTOT * HC * 4);
    float*    alpha = (float*)   carve((size_t)N_EDGES * HEADS * 4);
    unsigned* mEnc  = (unsigned*)carve((size_t)NTOT * HEADS * 4);
    float*    den   = (float*)   carve((size_t)NTOT * HEADS * 4);
    _Float16* wh    = (_Float16*)carve((size_t)512 * 1024 * 2);  // packed f16 weight staging

    // post-GNN buffers reuse q/k/v/alpha region (conv attention no longer live)
    char* post = (char*)qb; size_t poff = 0;
    auto pcarve = [&](size_t bytes) { size_t r = poff; poff = (poff + bytes + 255) & ~(size_t)255; return (void*)(post + r); };
    float*    g    = (float*)   pcarve((size_t)NTOT * EMBED * 4);
    _Float16* gh   = (_Float16*)pcarve((size_t)NTOT * EMBED * 2);
    float*    qkv  = (float*)   pcarve((size_t)NTOT * 3 * EMBED * 4);
    float*    oatt = (float*)   pcarve((size_t)NTOT * EMBED * 4);
    _Float16* oh   = (_Float16*)pcarve((size_t)NTOT * EMBED * 2);
    float*    o2   = (float*)   pcarve((size_t)NTOT * EMBED * 4);
    float*    lno  = (float*)   pcarve((size_t)NTOT * EMBED * 4);
    _Float16* comb = (_Float16*)pcarve((size_t)NTOT * 96 * 2);
    float*    ha   = (float*)   pcarve((size_t)NTOT * HIDDEN * 4);
    _Float16* hah  = (_Float16*)pcarve((size_t)NTOT * HIDDEN * 2);
    float*    hcb  = (float*)   pcarve((size_t)NTOT * HIDDEN * 4);

    float* probs = (float*)d_out;                    // [NTOT, 1024]
    float* vals  = probs + (size_t)NTOT * N_MISSIONS;// [NTOT, 1]

    const int TB = 256;
    auto gemm = [&](const _Float16* A, const _Float16* B, const float* bias,
                    float* C, int M, int N, int K, int relu) {
        dim3 block(128);                             // 4 waves/block
        if (N % 256 == 0) {                          // big GEMMs: 4 tiles/wave
            dim3 grid(M / 16, N / 256);
            wmma_gemm_kernel<4><<<grid, block, 0, stream>>>(A, B, bias, C, M, N, K, relu);
        } else {                                     // small projections
            dim3 grid(M / 16, N / 64);
            wmma_gemm_kernel<1><<<grid, block, 0, stream>>>(A, B, bias, C, M, N, K, relu);
        }
    };
    auto pack_w = [&](const float* w, int K0, int Kp, int N) {
        pack_w_kernel<<<cdiv(Kp * N, TB), TB, 0, stream>>>(w, wh, K0, Kp, N);
    };
    auto edge_pipeline = [&]() {
        fill_u32_kernel<<<cdiv(NTOT * HEADS, TB), TB, 0, stream>>>(mEnc, 0x007FFFFFu, NTOT * HEADS); // enc(-inf)
        fill_f32_kernel<<<cdiv(NTOT * HEADS, TB), TB, 0, stream>>>(den, 0.0f, NTOT * HEADS);
        int nt = N_EDGES * HEADS;
        edge_alpha_kernel  <<<cdiv(nt, TB), TB, 0, stream>>>(ei, qb, kb, alpha, mEnc);
        edge_exp_kernel    <<<cdiv(nt, TB), TB, 0, stream>>>(ei, alpha, mEnc, den);
        edge_scatter_kernel<<<cdiv(nt, TB), TB, 0, stream>>>(ei, alpha, den, vb, sbuf);
    };

    // ---- node features ----
    build_feat_kernel<<<cdiv(NTOT, TB), TB, 0, stream>>>(mc, amask, speeds, cm, at, feat);

    // ---- layer 1 (K=6, scalar GEMMs) ----
    {
        const float** P = (const float**)d_in;
        const float *wq = P[8], *bq = P[9], *wk = P[10], *bk = P[11];
        const float *wv = P[12], *bv = P[13], *wsm = P[14], *bs = P[15];
        int nel = NTOT * HC;
        gemm6_kernel<<<cdiv(nel, TB), TB, 0, stream>>>(feat, wq, bq, qb);
        gemm6_kernel<<<cdiv(nel, TB), TB, 0, stream>>>(feat, wk, bk, kb);
        gemm6_kernel<<<cdiv(nel, TB), TB, 0, stream>>>(feat, wv, bv, vb);
        gemm6_kernel<<<cdiv(nel, TB), TB, 0, stream>>>(feat, wsm, bs, sbuf);
        edge_pipeline();
        relu_to_half_kernel<<<cdiv(nel, TB), TB, 0, stream>>>(sbuf, xh, nel);
    }

    // ---- layers 2..4 (WMMA GEMMs, K=N=512) ----
    for (int l = 1; l < 4; ++l) {
        const float** P = (const float**)d_in;
        const float* W[4] = { P[8 + 8 * l + 0], P[8 + 8 * l + 2], P[8 + 8 * l + 4], P[8 + 8 * l + 6] };
        const float* B[4] = { P[8 + 8 * l + 1], P[8 + 8 * l + 3], P[8 + 8 * l + 5], P[8 + 8 * l + 7] };
        float* D[4] = { qb, kb, vb, sbuf };
        for (int j = 0; j < 4; ++j) {
            pack_w(W[j], HC, HC, HC);
            gemm(xh, wh, B[j], D[j], NTOT, HC, HC, 0);
        }
        edge_pipeline();
        relu_to_half_kernel<<<cdiv(NTOT * HC, TB), TB, 0, stream>>>(sbuf, xh, NTOT * HC);
    }

    // ---- gnn projection 512 -> 64 ----
    pack_w(gnn_w, HC, HC, EMBED);
    gemm(xh, wh, gnn_b, g, NTOT, EMBED, HC, 0);
    to_half_kernel<<<cdiv(NTOT * EMBED, TB), TB, 0, stream>>>(g, gh, NTOT * EMBED);

    // ---- per-graph MHA ----
    pack_w(ipw, EMBED, EMBED, 3 * EMBED);
    gemm(gh, wh, ipb, qkv, NTOT, 3 * EMBED, EMBED, 0);
    {
        dim3 grid(B_GRAPHS, HEADS), block(256);
        mha_kernel<<<grid, block, 0, stream>>>(qkv, oatt);
    }
    to_half_kernel<<<cdiv(NTOT * EMBED, TB), TB, 0, stream>>>(oatt, oh, NTOT * EMBED);
    pack_w(opw, EMBED, EMBED, EMBED);
    gemm(oh, wh, opb, o2, NTOT, EMBED, EMBED, 0);
    layernorm_kernel<<<cdiv(NTOT, TB), TB, 0, stream>>>(o2, lng, lnb, lno);

    // ---- heads ----
    build_comb_kernel<<<cdiv(NTOT, TB), TB, 0, stream>>>(lno, uavs, speeds, comb);

    // actor: 96(pad of 67) -> 128 relu -> 1024 softmax
    pack_w(aw1, EMBED + 3, 96, HIDDEN);
    gemm(comb, wh, ab1, ha, NTOT, HIDDEN, 96, 1);
    to_half_kernel<<<cdiv(NTOT * HIDDEN, TB), TB, 0, stream>>>(ha, hah, NTOT * HIDDEN);
    pack_w(aw2, HIDDEN, HIDDEN, N_MISSIONS);
    gemm(hah, wh, ab2, probs, NTOT, N_MISSIONS, HIDDEN, 0);
    softmax_rows_kernel<<<NTOT, 256, 0, stream>>>(probs);

    // critic: 96 -> 128 relu -> 1
    pack_w(cw1, EMBED + 3, 96, HIDDEN);
    gemm(comb, wh, cb1, hcb, NTOT, HIDDEN, 96, 1);
    critic_out_kernel<<<cdiv(NTOT, TB), TB, 0, stream>>>(hcb, cw2, cb2, vals);

    (void)in_sizes; (void)n_in; (void)out_size; (void)ws_size;
}
